// AwarenessCoreSNN_43284680409435
// MI455X (gfx1250) — matrix operand order repaired and checked by reading, not compiled, not run on hardware
//
#include <hip/hip_runtime.h>
#include <hip/hip_bf16.h>

// ---------------------------------------------------------------------------
// Types for CDNA5 WMMA (wave32): 16x16x32 bf16 -> f32
// ---------------------------------------------------------------------------
typedef __attribute__((ext_vector_type(16))) __bf16        v16bf;
typedef __attribute__((ext_vector_type(8)))  __bf16        v8bf;
typedef __attribute__((ext_vector_type(8)))  float         v8f;
typedef __attribute__((ext_vector_type(4)))  float         v4f;
typedef __attribute__((ext_vector_type(4)))  unsigned int  v4u;
typedef __attribute__((ext_vector_type(8)))  int           v8i;
typedef __attribute__((ext_vector_type(4)))  int           v4i;

// Problem constants
#define BATCH      16384
#define VIS        1024
#define TXT        768
#define CORE       192
#define NCLS       100
#define NCLS_PAD   112      // 7 n-tiles of 16
#define DECAY      0.95f
#define TH_CORE    0.5f
#define TH_CLS     0.5f

#define M_TILE     64       // rows per block
#define BLOCK      128      // 4 waves * 16 rows

__device__ __forceinline__ v8f wmma_bf16(v16bf a, v16bf b, v8f c) {
  return __builtin_amdgcn_wmma_f32_16x16x32_bf16(
      /*neg_a=*/false, a, /*neg_b=*/false, b,
      /*c_mod=*/(short)0, c, /*reuse_a=*/false, /*reuse_b=*/false);
}

// A-matrix 16x32 bf16 fragment from a global row-major f32 matrix.
__device__ __forceinline__ v16bf load_a_global(const float* __restrict__ h,
                                               int ld, int m, int k0, int lane) {
  const int off = (lane < 16) ? 0 : 16;
  const float* p = h + (size_t)m * ld + k0 + off;
  v4f a0 = *(const v4f*)(p + 0);
  v4f a1 = *(const v4f*)(p + 4);
  v4f a2 = *(const v4f*)(p + 8);
  v4f a3 = *(const v4f*)(p + 12);
  v16bf r;
#pragma unroll
  for (int i = 0; i < 4; ++i) {
    r[i]      = (__bf16)a0[i];
    r[4 + i]  = (__bf16)a1[i];
    r[8 + i]  = (__bf16)a2[i];
    r[12 + i] = (__bf16)a3[i];
  }
  return r;
}

// B-matrix 32x16 fragment from a pre-converted bf16 N x K row-major matrix:
// lane holds column n, 16 contiguous K values -> single 32B load, no cvt.
__device__ __forceinline__ v16bf load_b_bf16(const __bf16* __restrict__ W,
                                             int ld, int n, int k0, int lane) {
  return *(const v16bf*)(W + (size_t)n * ld + k0 + ((lane < 16) ? 0 : 16));
}

// LDS-sourced fragments (bf16 row-major [rows][ld])
__device__ __forceinline__ v16bf lds_a_frag(const __bf16* s, int ld,
                                            int m, int k0, int lane) {
  const __bf16* p = s + m * ld + k0 + ((lane < 16) ? 0 : 16);
  v8bf h0 = *(const v8bf*)(p);
  v8bf h1 = *(const v8bf*)(p + 8);
  v16bf r;
#pragma unroll
  for (int i = 0; i < 8; ++i) { r[i] = h0[i]; r[8 + i] = h1[i]; }
  return r;
}

__device__ __forceinline__ v16bf lds_b_frag(const __bf16* s, int ld,
                                            int n, int k0, int lane) {
  return *(const v16bf*)(s + n * ld + k0 + ((lane < 16) ? 0 : 16));
}

// ---------------------------------------------------------------------------
// TDM: build a Tensor DMA Descriptor (D#) and issue TENSOR_LOAD_TO_LDS.
// 2D tensor: dim0 = row length, dim1 = #rows; tile (tile0 x tile1) starting
// at gaddr. data_size_code: 0=1B 1=2B 2=4B 3=8B. (CDNA5 ISA)
// ---------------------------------------------------------------------------
__device__ __forceinline__ void tdm_load_2d(unsigned lds_off, const void* gaddr,
                                            unsigned dim0, unsigned dim1,
                                            unsigned tile0, unsigned tile1,
                                            unsigned stride0,
                                            unsigned data_size_code) {
  const unsigned long long ga = (unsigned long long)gaddr;
  v4u g0;
  g0[0] = 1u;                                               // count=1, user mode
  g0[1] = lds_off;                                          // lds_addr
  g0[2] = (unsigned)ga;                                     // global_addr[31:0]
  g0[3] = ((unsigned)(ga >> 32) & 0x01FFFFFFu) | (2u << 30); // addr[56:32]|type=2
  v8i g1;
  g1[0] = (int)(data_size_code << 16);                      // mask=0, data_size
  g1[1] = (int)((dim0 & 0xFFFFu) << 16);                    // tensor_dim0[15:0]
  g1[2] = (int)(((dim0 >> 16) & 0xFFFFu) | ((dim1 & 0xFFFFu) << 16));
  g1[3] = (int)(((dim1 >> 16) & 0xFFFFu) | ((tile0 & 0xFFFFu) << 16));
  g1[4] = (int)(tile1 & 0xFFFFu);                           // tile_dim1, tile_dim2=0
  g1[5] = (int)stride0;                                     // tensor_dim0_stride lo
  g1[6] = 0;
  g1[7] = 0;
  const v4i z4 = (v4i){0, 0, 0, 0};                         // groups 2/3 unused (2D)
#if __clang_major__ >= 23
  const v8i z8 = (v8i){0, 0, 0, 0, 0, 0, 0, 0};
  __builtin_amdgcn_tensor_load_to_lds(g0, g1, z4, z4, z8, 0);
#else
  __builtin_amdgcn_tensor_load_to_lds(g0, g1, z4, z4, 0);
#endif
}

// ---------------------------------------------------------------------------
// Kernel 0: one-shot weight conversion f32 -> bf16 into workspace.
// Concatenated ranges: Wvis | Wtxt | Wrec | Wcls (padded to 112 rows).
// ---------------------------------------------------------------------------
#define NVIS_W  (CORE * VIS)
#define NTXT_W  (CORE * TXT)
#define NREC_W  (CORE * CORE)
#define NCLS_W  (NCLS_PAD * CORE)
#define NCONV   (NVIS_W + NTXT_W + NREC_W + NCLS_W)

__global__ void convert_weights(const float* __restrict__ Wvis,
                                const float* __restrict__ Wtxt,
                                const float* __restrict__ Wrec,
                                const float* __restrict__ Wcls,
                                __bf16* __restrict__ wvis,
                                __bf16* __restrict__ wtxt,
                                __bf16* __restrict__ wrec,
                                __bf16* __restrict__ wcls) {
  int i = blockIdx.x * blockDim.x + threadIdx.x;
  if (i < NVIS_W) { wvis[i] = (__bf16)Wvis[i]; return; }
  i -= NVIS_W;
  if (i < NTXT_W) { wtxt[i] = (__bf16)Wtxt[i]; return; }
  i -= NTXT_W;
  if (i < NREC_W) { wrec[i] = (__bf16)Wrec[i]; return; }
  i -= NREC_W;
  if (i < NCLS_W) {
    const int r = i / CORE, c = i - r * CORE;
    wcls[i] = (__bf16)((r < NCLS) ? Wcls[r * CORE + c] : 0.0f);
  }
}

// ---------------------------------------------------------------------------
// Kernel 1: drive = h_vis @ W_vis.T + h_text @ W_text.T ; emits z_vis, z_text.
// One wave per 16 output rows, 12 n-tiles of N=192. B from bf16 weights.
// ---------------------------------------------------------------------------
__global__ void drive_kernel(const float* __restrict__ hvis,
                             const float* __restrict__ htxt,
                             const __bf16* __restrict__ wvis,
                             const __bf16* __restrict__ wtxt,
                             float* __restrict__ z_vis,
                             float* __restrict__ z_text,
                             float* __restrict__ drive) {
  const int lane  = threadIdx.x & 31;
  const int wave  = threadIdx.x >> 5;
  const int mbase = blockIdx.x * M_TILE + wave * 16;
  const int arow  = mbase + (lane & 15);
  const int rbase = mbase + ((lane < 16) ? 0 : 8);
  const int col   = lane & 15;

  v8f acc[12];
#pragma unroll
  for (int nt = 0; nt < 12; ++nt) acc[nt] = (v8f){0,0,0,0,0,0,0,0};

  // ---- vis projection ----
  for (int k0 = 0; k0 < VIS; k0 += 32) {
    if (k0 + 64 < VIS)
      __builtin_prefetch(hvis + (size_t)arow * VIS + k0 + 64, 0, 0);
    v16bf a = load_a_global(hvis, VIS, arow, k0, lane);
#pragma unroll
    for (int nt = 0; nt < 12; nt += 2) {  // paired to let loads overlap
      v16bf b0 = load_b_bf16(wvis, VIS, (nt    ) * 16 + col, k0, lane);
      v16bf b1 = load_b_bf16(wvis, VIS, (nt + 1) * 16 + col, k0, lane);
      acc[nt]     = wmma_bf16(a, b0, acc[nt]);
      acc[nt + 1] = wmma_bf16(a, b1, acc[nt + 1]);
    }
  }
#pragma unroll
  for (int nt = 0; nt < 12; ++nt) {
    const int n = nt * 16 + col;
#pragma unroll
    for (int r = 0; r < 8; ++r)
      z_vis[(size_t)(rbase + r) * CORE + n] = acc[nt][r];
  }

  // ---- text projection (reuse accumulators) ----
#pragma unroll
  for (int nt = 0; nt < 12; ++nt) acc[nt] = (v8f){0,0,0,0,0,0,0,0};
  for (int k0 = 0; k0 < TXT; k0 += 32) {
    v16bf a = load_a_global(htxt, TXT, arow, k0, lane);
#pragma unroll
    for (int nt = 0; nt < 12; nt += 2) {
      v16bf b0 = load_b_bf16(wtxt, TXT, (nt    ) * 16 + col, k0, lane);
      v16bf b1 = load_b_bf16(wtxt, TXT, (nt + 1) * 16 + col, k0, lane);
      acc[nt]     = wmma_bf16(a, b0, acc[nt]);
      acc[nt + 1] = wmma_bf16(a, b1, acc[nt + 1]);
    }
  }
#pragma unroll
  for (int nt = 0; nt < 12; ++nt) {
    const int n = nt * 16 + col;
#pragma unroll
    for (int r = 0; r < 8; ++r) {
      const size_t idx = (size_t)(rbase + r) * CORE + n;
      const float zt = acc[nt][r];
      z_text[idx] = zt;
      drive[idx]  = zt + z_vis[idx];  // same-wave store->load: kept in order
    }
  }
}

// ---------------------------------------------------------------------------
// Kernel 2: fused SNN recurrence, barrier-free time loop (wave-private rows).
// All operands staged into LDS by the Tensor Data Mover.
//   sWrec [192][192] bf16 | sWcls [112][192] bf16 | sDrive [64][192] f32
//   sV ping-pong [2][64][192] bf16 | sSpk [64][192] bf16 | sZ [64][192] f32
// ---------------------------------------------------------------------------
#define OFF_WREC   0
#define OFF_WCLS   (OFF_WREC + CORE * CORE * 2)            // 73728
#define OFF_DRIVE  (OFF_WCLS + NCLS_PAD * CORE * 2)        // 116736
#define OFF_V      (OFF_DRIVE + M_TILE * CORE * 4)         // 165888
#define OFF_SPK    (OFF_V + 2 * M_TILE * CORE * 2)         // 215040
#define OFF_Z      (OFF_SPK + M_TILE * CORE * 2)           // 239616
#define LDS_BYTES  (OFF_Z + M_TILE * CORE * 4)             // 288768

__global__ void snn_kernel(const __bf16* __restrict__ wrec_bf,
                           const __bf16* __restrict__ wcls_bf,
                           const float* __restrict__ drive_g,
                           const int*   __restrict__ tsteps,
                           float* __restrict__ z_core,
                           float* __restrict__ cls_out) {
  extern __shared__ char smem[];
  __bf16* sWrec  = (__bf16*)(smem + OFF_WREC);
  __bf16* sWcls  = (__bf16*)(smem + OFF_WCLS);
  float*  sDrive = (float*)(smem + OFF_DRIVE);
  __bf16* sV     = (__bf16*)(smem + OFF_V);
  __bf16* sSpk   = (__bf16*)(smem + OFF_SPK);
  float*  sZ     = (float*)(smem + OFF_Z);

  const int tid       = threadIdx.x;
  const int mbase_blk = blockIdx.x * M_TILE;

  // ---- TDM DMA: drive tile + both weight matrices straight into LDS ----
  if (tid < 32) {  // one wave issues; TDM ignores EXEC
    tdm_load_2d((unsigned)(size_t)sDrive,
                drive_g + (size_t)mbase_blk * CORE,
                CORE, BATCH, CORE, M_TILE, CORE, /*f32*/2u);
    tdm_load_2d((unsigned)(size_t)sWrec, wrec_bf,
                CORE * CORE, 1, CORE * CORE, 1, CORE * CORE, /*bf16*/1u);
    tdm_load_2d((unsigned)(size_t)sWcls, wcls_bf,
                NCLS_PAD * CORE, 1, NCLS_PAD * CORE, 1, NCLS_PAD * CORE, 1u);
  }

  // ---- init v_core ping-pong + spike sums ----
  for (int i = tid; i < 2 * M_TILE * CORE; i += BLOCK) sV[i] = (__bf16)0.0f;
  for (int i = tid; i < M_TILE * CORE; i += BLOCK) sZ[i] = 0.0f;

  __builtin_amdgcn_s_wait_tensorcnt(0);
  __syncthreads();

  const int lane = tid & 31;
  const int wave = tid >> 5;
  const int col  = lane & 15;
  const int rloc = wave * 16 + ((lane < 16) ? 0 : 8);   // C/D local row base
  const int arow = wave * 16 + col;                     // A local row
  const int T    = tsteps[0];

  v8f vcore[12];
  v8f vcls[7];
#pragma unroll
  for (int nt = 0; nt < 12; ++nt) vcore[nt] = (v8f){0,0,0,0,0,0,0,0};
#pragma unroll
  for (int ct = 0; ct < 7; ++ct)  vcls[ct]  = (v8f){0,0,0,0,0,0,0,0};

  int cur = 0;
  for (int t = 0; t < T; ++t) {
    const __bf16* sVcur = sV + cur * (M_TILE * CORE);
    __bf16*       sVnxt = sV + (cur ^ 1) * (M_TILE * CORE);

    // core layer: acc = drive + v_core_prev @ W_rec.T  (C seeded with drive)
#pragma unroll
    for (int nt = 0; nt < 12; ++nt) {
      v8f acc;
#pragma unroll
      for (int r = 0; r < 8; ++r)
        acc[r] = sDrive[(rloc + r) * CORE + nt * 16 + col];
#pragma unroll
      for (int kt = 0; kt < 6; ++kt) {
        v16bf a = lds_a_frag(sVcur, CORE, arow, kt * 32, lane);
        v16bf b = lds_b_frag(sWrec, CORE, nt * 16 + col, kt * 32, lane);
        acc = wmma_bf16(a, b, acc);
      }
#pragma unroll
      for (int r = 0; r < 8; ++r) {
        const float vv  = DECAY * vcore[nt][r] + acc[r];
        const float spk = (vv >= TH_CORE) ? 1.0f : 0.0f;
        const float vn  = vv * (1.0f - spk);
        vcore[nt][r] = vn;
        const int idx = (rloc + r) * CORE + nt * 16 + col;
        sVnxt[idx] = (__bf16)vn;
        sSpk[idx]  = (__bf16)spk;
        sZ[idx]   += spk;
      }
    }

    // class layer: core_spk @ W_cls.T (same-wave LDS producer->consumer)
#pragma unroll
    for (int ct = 0; ct < 7; ++ct) {
      v8f acc = (v8f){0,0,0,0,0,0,0,0};
#pragma unroll
      for (int kt = 0; kt < 6; ++kt) {
        v16bf a = lds_a_frag(sSpk, CORE, arow, kt * 32, lane);
        v16bf b = lds_b_frag(sWcls, CORE, ct * 16 + col, kt * 32, lane);
        acc = wmma_bf16(a, b, acc);
      }
      const int n = ct * 16 + col;
#pragma unroll
      for (int r = 0; r < 8; ++r) {
        const float vv  = DECAY * vcls[ct][r] + acc[r];
        const float spk = (vv >= TH_CLS) ? 1.0f : 0.0f;
        vcls[ct][r] = vv * (1.0f - spk);
        if (n < NCLS) {
          const size_t grow = (size_t)(mbase_blk + rloc + r);
          cls_out[(size_t)t * BATCH * NCLS + grow * NCLS + n] = spk;
        }
      }
    }
    cur ^= 1;
  }

  __syncthreads();
  const float invT = 1.0f / (float)T;
  for (int i = tid; i < M_TILE * CORE; i += BLOCK)
    z_core[(size_t)mbase_blk * CORE + i] = sZ[i] * invT;
}

// ---------------------------------------------------------------------------
extern "C" void kernel_launch(void* const* d_in, const int* in_sizes, int n_in,
                              void* d_out, int out_size, void* d_ws, size_t ws_size,
                              hipStream_t stream) {
  const float* hvis  = (const float*)d_in[0];
  const float* htxt  = (const float*)d_in[1];
  const float* Wvis  = (const float*)d_in[2];
  const float* Wtxt  = (const float*)d_in[3];
  const float* Wrec  = (const float*)d_in[4];
  const float* Wcls  = (const float*)d_in[5];
  const int*   tstep = (const int*)d_in[6];
  (void)in_sizes; (void)n_in; (void)out_size; (void)ws_size;

  float* out    = (float*)d_out;
  float* z_core = out;
  float* z_vis  = out + (size_t)BATCH * CORE;
  float* z_text = out + (size_t)2 * BATCH * CORE;
  float* cls    = out + (size_t)3 * BATCH * CORE;

  // Workspace layout: drive f32 | Wvis bf16 | Wtxt bf16 | Wrec bf16 | Wcls bf16
  float*  drive   = (float*)d_ws;
  __bf16* wvis_bf = (__bf16*)(drive + (size_t)BATCH * CORE);
  __bf16* wtxt_bf = wvis_bf + NVIS_W;
  __bf16* wrec_bf = wtxt_bf + NTXT_W;
  __bf16* wcls_bf = wrec_bf + NREC_W;

  static_assert(LDS_BYTES <= 320 * 1024, "LDS over WGP budget");
  (void)hipFuncSetAttribute((const void*)snn_kernel,
                            hipFuncAttributeMaxDynamicSharedMemorySize,
                            LDS_BYTES);

  dim3 blk(BLOCK);
  dim3 grd(BATCH / M_TILE);
  convert_weights<<<dim3(NCONV / 256), dim3(256), 0, stream>>>(
      Wvis, Wtxt, Wrec, Wcls, wvis_bf, wtxt_bf, wrec_bf, wcls_bf);
  drive_kernel<<<grd, blk, 0, stream>>>(hvis, htxt, wvis_bf, wtxt_bf,
                                        z_vis, z_text, drive);
  snn_kernel<<<grd, blk, LDS_BYTES, stream>>>(wrec_bf, wcls_bf, drive, tstep,
                                              z_core, cls);
}
